// CausalSelfAttention_82042465289080
// MI455X (gfx1250) — compile-verified, matrix operand order checked by reading, not run on hardware
//
#include <hip/hip_runtime.h>

// ---------------------------------------------------------------------------
// CDNA5 (gfx1250, wave32) fused causal self-attention.
//   B=4, S=2048, E=1024, H=16, Dh=64.  All matmuls via v_wmma_f32_16x16x32_bf16.
// ---------------------------------------------------------------------------

typedef __attribute__((ext_vector_type(16))) __bf16 v16bf;
typedef __attribute__((ext_vector_type(8)))  __bf16 v8bf;
typedef __attribute__((ext_vector_type(4)))  __bf16 v4bf;
typedef __attribute__((ext_vector_type(8)))  float  v8f;

#define BATCH 4
#define SEQ   2048
#define EMBED 1024
#define HEADS 16
#define HDIM  64
#define MTOT  (BATCH * SEQ)          // 8192 rows
#define LOG2E 1.44269504088896340f
// 1/sqrt(Dh) * log2(e), folded into the bf16 Q tensor at projection time so
// the score WMMA directly produces log2-domain logits.
#define QSCALE (0.125f * LOG2E)

__device__ __forceinline__ v8f zero8() {
  v8f z = {0.f, 0.f, 0.f, 0.f, 0.f, 0.f, 0.f, 0.f};
  return z;
}

// Wave-uniform value -> SGPR (scalar loop control, saddr-form addressing).
__device__ __forceinline__ int uniform(int v) {
#if __has_builtin(__builtin_amdgcn_readfirstlane)
  return __builtin_amdgcn_readfirstlane(v);
#else
  return v;
#endif
}

__device__ __forceinline__ float fast_exp2(float x) {
#if __has_builtin(__builtin_amdgcn_exp2f)
  return __builtin_amdgcn_exp2f(x);   // raw v_exp_f32; args are <= 0, bounded
#else
  return exp2f(x);
#endif
}

// --- 16x32 bf16 fragment loader (A operand; B operand uses the same pattern
//     on a [N][K]-stored matrix: lane = row/col index, K contiguous in lane).
// Per ISA 7.12.2 (16-bit A 16x32): lanes 0-15 hold row=lane with K=0..7 in
// V0-3 and K=16..23 in V4-7; lanes 16-31 hold row=lane-16 with K=8..15 /
// K=24..31.  Two 16-byte loads per lane.
__device__ __forceinline__ v16bf load_tile(const __bf16* base, int ld) {
  int lane = threadIdx.x & 31;
  int half = lane >> 4;
  int r    = lane & 15;
  const __bf16* p = base + (size_t)r * ld + half * 8;
  struct Halves { v8bf lo; v8bf hi; };
  union { v16bf v; Halves h; } u;
  u.h.lo = *(const v8bf*)(p);        // K = half*8 + 0..7
  u.h.hi = *(const v8bf*)(p + 16);   // K = 16 + half*8 + 0..7
  return u.v;
}

__device__ __forceinline__ v8f wmma_bf16(v16bf a, v16bf b, v8f c) {
  // D = A(16x32) x B(32x16) + C(16x16 f32)
  return __builtin_amdgcn_wmma_f32_16x16x32_bf16(
      /*neg_a=*/false, a, /*neg_b=*/false, b,
      /*c_mod=*/(short)0, c, /*reuse_a=*/false, /*reuse_b=*/false);
}

// 16-lane-half xor-butterfly max via v_permlane16_b32 (pure VALU, no DS).
// C/D layout puts one score row in one VGPR of one 16-lane half, so this
// reduces exactly one row; both halves reduce their own row simultaneously.
__device__ __forceinline__ float lanexor(float v, unsigned lo, unsigned hi) {
#if __has_builtin(__builtin_amdgcn_permlane16)
  int i = __float_as_int(v);
  i = __builtin_amdgcn_permlane16(i, i, lo, hi, false, false);
  return __int_as_float(i);
#else
  (void)lo; (void)hi;
  return v;  // unused in fallback path
#endif
}
__device__ __forceinline__ float half16_max(float v) {
#if __has_builtin(__builtin_amdgcn_permlane16)
  v = fmaxf(v, lanexor(v, 0x67452301u, 0xEFCDAB89u)); // lane ^= 1
  v = fmaxf(v, lanexor(v, 0x54761032u, 0xDCFE98BAu)); // lane ^= 2
  v = fmaxf(v, lanexor(v, 0x32107654u, 0xBA98FEDCu)); // lane ^= 4
  v = fmaxf(v, lanexor(v, 0xFEDCBA98u, 0x76543210u)); // lane ^= 8
#else
  for (int m = 1; m < 16; m <<= 1) v = fmaxf(v, __shfl_xor(v, m, 32));
#endif
  return v;
}

// ---------------------------------------------------------------------------
// Kernel 1: fp32 -> bf16 (x matrix), 4 elements/thread
// ---------------------------------------------------------------------------
__global__ void cvt_f32_bf16(const float* __restrict__ in,
                             __bf16* __restrict__ out, int n) {
  int i = (blockIdx.x * blockDim.x + threadIdx.x) * 4;
  if (i >= n) return;
  float4 f = *(const float4*)(in + i);
  v4bf o4 = {(__bf16)f.x, (__bf16)f.y, (__bf16)f.z, (__bf16)f.w};
  *(v4bf*)(out + i) = o4;
}

// ---------------------------------------------------------------------------
// Kernel 2: W[1024][1024] fp32 -> Wt[n][k] bf16 (transpose via LDS)
// ---------------------------------------------------------------------------
__global__ void transposeW(const float* __restrict__ W, __bf16* __restrict__ Wt) {
  __shared__ float t[16][17];
  int k = blockIdx.y * 16 + threadIdx.y;
  int n = blockIdx.x * 16 + threadIdx.x;
  t[threadIdx.y][threadIdx.x] = W[(size_t)k * EMBED + n];
  __syncthreads();
  int nn = blockIdx.x * 16 + threadIdx.y;
  int kk = blockIdx.y * 16 + threadIdx.x;
  Wt[(size_t)nn * EMBED + kk] = (__bf16)t[threadIdx.x][threadIdx.y];
}

// ---------------------------------------------------------------------------
// Kernel 3: generic WMMA GEMM, one wave = 16(M) x 64(N) tile, K-loop of 32.
//   A: [M][K] bf16 row-major.  Bt: [N][K] bf16 (pre-transposed weights).
// MODE 0: fp32 [M][N] plain      (final projection, writes d_out)
// MODE 1: bf16 [B,H,S,Dh]        (K)
// MODE 2: bf16 [B,H,Dh,S]        (V transposed: P@V B-operand is K-contiguous)
// MODE 3: bf16 [B,H,S,Dh]*QSCALE (Q with 1/sqrt(Dh)*log2e pre-folded)
// ---------------------------------------------------------------------------
template <int MODE>
__global__ void gemm_wmma_kernel(const __bf16* __restrict__ A,
                                 const __bf16* __restrict__ Bt,
                                 void* __restrict__ out,
                                 int M, int N, int K) {
  int w       = uniform((blockIdx.x * blockDim.x + threadIdx.x) >> 5);
  int nStrips = N >> 6;
  int mTile   = w / nStrips;
  int nStrip  = w % nStrips;
  if (mTile * 16 >= M) return;

  const __bf16* Arow = A  + (size_t)mTile * 16 * K;
  const __bf16* B0   = Bt + (size_t)nStrip * 64 * K;

  v8f acc[4];
#pragma unroll
  for (int nt = 0; nt < 4; ++nt) acc[nt] = zero8();

  for (int k0 = 0; k0 < K; k0 += 32) {
    __builtin_prefetch(Arow + k0 + 128, 0, 0);   // global_prefetch_b8
    v16bf a = load_tile(Arow + k0, K);
#pragma unroll
    for (int nt = 0; nt < 4; ++nt) {
      v16bf bfrag = load_tile(B0 + (size_t)nt * 16 * K + k0, K);
      acc[nt] = wmma_bf16(a, bfrag, acc[nt]);
    }
  }

  int lane = threadIdx.x & 31, col = lane & 15, hi = lane >> 4;
#pragma unroll
  for (int nt = 0; nt < 4; ++nt) {
#pragma unroll
    for (int r = 0; r < 8; ++r) {
      int m = mTile * 16 + r + hi * 8;       // C/D layout: row = r + 8*(lane>=16)
      int n = nStrip * 64 + nt * 16 + col;   //             col = lane&15
      float val = acc[nt][r];
      if constexpr (MODE == 0) {
        ((float*)out)[(size_t)m * N + n] = val;
      } else if constexpr (MODE == 1 || MODE == 3) {
        if constexpr (MODE == 3) val *= QSCALE;
        int b = m >> 11, s = m & 2047, hh = n >> 6, d = n & 63;
        ((__bf16*)out)[((((size_t)b * HEADS + hh) * SEQ + s) << 6) + d] = (__bf16)val;
      } else {
        int b = m >> 11, s = m & 2047, hh = n >> 6, d = n & 63;
        ((__bf16*)out)[(((size_t)b * HEADS + hh) * HDIM + d) * SEQ + s] = (__bf16)val;
      }
    }
  }
}

// ---------------------------------------------------------------------------
// Flash-attention per-wave state (kept in registers via SROA/inlining)
// ---------------------------------------------------------------------------
struct AttnState {
  v16bf qf[2][2];     // 2 sub-tiles x 2 K-steps of Q
  float mrow[2][8];   // running row max (log2 domain)
  v8f   lacc[2];      // running row sums (WMMA D layout, lane-broadcast)
  v8f   o[2][4];      // 32x64 fp32 output accumulators
};

// One 32-wide kv step.  MASKED=true only for the single diagonal tile
// (kv0 == q0); all bulk steps skip every causal compare/select.
template <bool MASKED>
__device__ __forceinline__ void attn_step(
    AttnState& st, int kv0, int q0, int col, int hi,
    const __bf16* __restrict__ Kb, const __bf16* __restrict__ Vb,
    __bf16* pb /* this wave's LDS P buffer: 2 x 16x32 bf16 */, v16bf ones) {
  // ---- K fragments, shared by both q sub-tiles (die before V loads) ----
  v16bf kf0 = load_tile(Kb + (size_t)(kv0     ) * HDIM + 0,  HDIM);
  v16bf kf1 = load_tile(Kb + (size_t)(kv0     ) * HDIM + 32, HDIM);
  v16bf kf2 = load_tile(Kb + (size_t)(kv0 + 16) * HDIM + 0,  HDIM);
  v16bf kf3 = load_tile(Kb + (size_t)(kv0 + 16) * HDIM + 32, HDIM);

  // ---- per sub-tile: scores -> (mask) -> online softmax -> P to LDS ----
#pragma unroll
  for (int qi = 0; qi < 2; ++qi) {
    v8f s0, s1;
    s0 = wmma_bf16(st.qf[qi][0], kf0, zero8());
    s0 = wmma_bf16(st.qf[qi][1], kf1, s0);
    s1 = wmma_bf16(st.qf[qi][0], kf2, zero8());
    s1 = wmma_bf16(st.qf[qi][1], kf3, s1);
#pragma unroll
    for (int r = 0; r < 8; ++r) {
      float a0, a1;
      if constexpr (MASKED) {
        int row = q0 + qi * 16 + r + hi * 8;
        int c0  = kv0 + col, c1 = c0 + 16;
        a0 = (c0 <= row) ? s0[r] : -INFINITY;
        a1 = (c1 <= row) ? s1[r] : -INFINITY;
      } else {
        a0 = s0[r];
        a1 = s1[r];
      }
      float mnew  = fmaxf(st.mrow[qi][r], half16_max(fmaxf(a0, a1)));
      float alpha = fast_exp2(st.mrow[qi][r] - mnew);  // -inf -> 0 first step
      float p0 = fast_exp2(a0 - mnew);                 // masked -> 0
      float p1 = fast_exp2(a1 - mnew);
      st.mrow[qi][r] = mnew;
      st.lacc[qi][r] *= alpha;
#pragma unroll
      for (int dt = 0; dt < 4; ++dt) st.o[qi][dt][r] *= alpha;
      int prow = r + hi * 8;
      pb[qi * 512 + prow * 32 + col]      = (__bf16)p0;
      pb[qi * 512 + prow * 32 + col + 16] = (__bf16)p1;
    }
  }

  // cross-lane transpose via LDS: drain DS stores before A-fragment reload
  asm volatile("s_wait_dscnt 0" ::: "memory");

  // ---- V fragments, shared by both q sub-tiles ----
  v16bf vf0 = load_tile(Vb + (size_t)0  * SEQ + kv0, SEQ);
  v16bf vf1 = load_tile(Vb + (size_t)16 * SEQ + kv0, SEQ);
  v16bf vf2 = load_tile(Vb + (size_t)32 * SEQ + kv0, SEQ);
  v16bf vf3 = load_tile(Vb + (size_t)48 * SEQ + kv0, SEQ);

#pragma unroll
  for (int qi = 0; qi < 2; ++qi) {
    v16bf pa = load_tile(pb + qi * 512, 32);     // P as 16x32 A operand
    st.lacc[qi] = wmma_bf16(pa, ones, st.lacc[qi]);  // row sums, broadcast
    st.o[qi][0] = wmma_bf16(pa, vf0, st.o[qi][0]);   // O += P . V
    st.o[qi][1] = wmma_bf16(pa, vf1, st.o[qi][1]);
    st.o[qi][2] = wmma_bf16(pa, vf2, st.o[qi][2]);
    st.o[qi][3] = wmma_bf16(pa, vf3, st.o[qi][3]);
  }
}

// ---------------------------------------------------------------------------
// Kernel 4: causal flash attention. One wave owns a 32-row query tile (two
// 16-row sub-tiles sharing K/V fragments -> 18 WMMAs per 16 b128 loads per
// kv-step).  Bulk kv-steps (kv0 < q0) are entirely below the diagonal and
// skip all causal compares; only the final diagonal tile is masked.
// Row-max: v_permlane16 butterflies (VALU only).  Row-sum: one extra WMMA
// against an all-ones B fragment (D layout broadcasts the sum to all lanes).
// Wave-uniform ids forced into SGPRs for scalar loop control/addressing.
//   Q: [B,H,S,Dh] bf16 pre-scaled by QSCALE (log2-domain logits).
//   K: [B,H,S,Dh] bf16.  Vt: [B,H,Dh,S] bf16.  comb: [B,S,H*Dh] bf16.
// ---------------------------------------------------------------------------
__global__ void __launch_bounds__(256, 1)
flash_attn_kernel(const __bf16* __restrict__ Q,
                  const __bf16* __restrict__ Kg,
                  const __bf16* __restrict__ Vt,
                  __bf16* __restrict__ comb) {
  __shared__ __bf16 pbuf[8][2][16 * 32];       // 8 waves x 2 sub-tiles x 1KB
  int wib  = uniform(threadIdx.x >> 5);
  int lane = threadIdx.x & 31;
  int w    = uniform(blockIdx.x * 8 + wib);    // 4096 waves total
  int bh   = w >> 6;                           // 0..63  (batch*head)
  int qt   = w & 63;                           // 0..63  (32-row query tile)
  int b    = bh >> 4, h = bh & 15;
  int q0   = qt * 32;

  const __bf16* Qb = Q  + (size_t)bh * SEQ * HDIM;
  const __bf16* Kb = Kg + (size_t)bh * SEQ * HDIM;
  const __bf16* Vb = Vt + (size_t)bh * HDIM * SEQ;
  __bf16* pb = &pbuf[wib][0][0];

  AttnState st;
#pragma unroll
  for (int qi = 0; qi < 2; ++qi) {
    st.qf[qi][0] = load_tile(Qb + (size_t)(q0 + qi * 16) * HDIM + 0,  HDIM);
    st.qf[qi][1] = load_tile(Qb + (size_t)(q0 + qi * 16) * HDIM + 32, HDIM);
    st.lacc[qi] = zero8();
#pragma unroll
    for (int r = 0; r < 8; ++r) st.mrow[qi][r] = -INFINITY;
#pragma unroll
    for (int dt = 0; dt < 4; ++dt) st.o[qi][dt] = zero8();
  }

  v16bf ones;
#pragma unroll
  for (int i = 0; i < 16; ++i) ones[i] = (__bf16)1.0f;

  int col = lane & 15, hi = lane >> 4;

  // bulk phase: strictly below the diagonal, no causal compares at all
  for (int kv0 = 0; kv0 < q0; kv0 += 32)
    attn_step<false>(st, kv0, q0, col, hi, Kb, Vb, pb, ones);
  // diagonal tile: per-row causal mask
  attn_step<true>(st, q0, q0, col, hi, Kb, Vb, pb, ones);

  // ---- epilogue: O /= l, write combined [B,S,H*Dh] bf16 ----
#pragma unroll
  for (int qi = 0; qi < 2; ++qi) {
#pragma unroll
    for (int r = 0; r < 8; ++r) {
      float inv = 1.0f / st.lacc[qi][r];
      int row = q0 + qi * 16 + r + hi * 8;
      size_t base = ((size_t)b * SEQ + row) * EMBED + h * HDIM;
#pragma unroll
      for (int dt = 0; dt < 4; ++dt)
        comb[base + dt * 16 + col] = (__bf16)(st.o[qi][dt][r] * inv);
    }
  }
}

// ---------------------------------------------------------------------------
// Workspace layout (bytes, 256B aligned)
// ---------------------------------------------------------------------------
#define SZ_XB   ((size_t)MTOT * EMBED * 2)            // 16 MiB
#define SZ_WT   ((size_t)EMBED * EMBED * 2)           // 2 MiB each
#define OFF_XB   ((size_t)0)
#define OFF_WQT  (OFF_XB  + SZ_XB)
#define OFF_WKT  (OFF_WQT + SZ_WT)
#define OFF_WVT  (OFF_WKT + SZ_WT)
#define OFF_WOT  (OFF_WVT + SZ_WT)
#define OFF_Q    (OFF_WOT + SZ_WT)
#define OFF_K    (OFF_Q   + SZ_XB)
#define OFF_VT   (OFF_K   + SZ_XB)
#define OFF_COMB (OFF_VT  + SZ_XB)

extern "C" void kernel_launch(void* const* d_in, const int* in_sizes, int n_in,
                              void* d_out, int out_size, void* d_ws, size_t ws_size,
                              hipStream_t stream) {
  const float* x  = (const float*)d_in[0];
  const float* Wq = (const float*)d_in[1];
  const float* Wk = (const float*)d_in[2];
  const float* Wv = (const float*)d_in[3];
  const float* Wo = (const float*)d_in[4];

  char* ws = (char*)d_ws;
  __bf16* xb   = (__bf16*)(ws + OFF_XB);
  __bf16* wqt  = (__bf16*)(ws + OFF_WQT);
  __bf16* wkt  = (__bf16*)(ws + OFF_WKT);
  __bf16* wvt  = (__bf16*)(ws + OFF_WVT);
  __bf16* wot  = (__bf16*)(ws + OFF_WOT);
  __bf16* qb   = (__bf16*)(ws + OFF_Q);
  __bf16* kb   = (__bf16*)(ws + OFF_K);
  __bf16* vtb  = (__bf16*)(ws + OFF_VT);
  __bf16* comb = (__bf16*)(ws + OFF_COMB);

  // 1) precision conversion + weight transposes
  cvt_f32_bf16<<<(MTOT * EMBED / 4 + 255) / 256, 256, 0, stream>>>(x, xb, MTOT * EMBED);
  dim3 tg(EMBED / 16, EMBED / 16), tb(16, 16);
  transposeW<<<tg, tb, 0, stream>>>(Wq, wqt);
  transposeW<<<tg, tb, 0, stream>>>(Wk, wkt);
  transposeW<<<tg, tb, 0, stream>>>(Wv, wvt);
  transposeW<<<tg, tb, 0, stream>>>(Wo, wot);

  // 2) QKV projections (WMMA). waves = (8192/16)*(1024/64) = 8192 -> 1024 blocks
  gemm_wmma_kernel<3><<<1024, 256, 0, stream>>>(xb, wqt, qb,  MTOT, EMBED, EMBED);
  gemm_wmma_kernel<1><<<1024, 256, 0, stream>>>(xb, wkt, kb,  MTOT, EMBED, EMBED);
  gemm_wmma_kernel<2><<<1024, 256, 0, stream>>>(xb, wvt, vtb, MTOT, EMBED, EMBED);

  // 3) causal flash attention: 64 bh * 64 q-tiles = 4096 waves -> 512 blocks
  flash_attn_kernel<<<512, 256, 0, stream>>>(qb, kb, vtb, comb);

  // 4) output projection -> fp32 d_out
  gemm_wmma_kernel<0><<<1024, 256, 0, stream>>>(comb, wot, d_out, MTOT, EMBED, EMBED);
}
